// LSTMNetwork_42743514530045
// MI455X (gfx1250) — matrix operand order repaired and checked by reading, not compiled, not run on hardware
//
#include <hip/hip_runtime.h>
#include <hip/hip_bf16.h>

typedef __attribute__((ext_vector_type(16))) __bf16       v16bf;
typedef __attribute__((ext_vector_type(8)))  float        v8f;
typedef __attribute__((ext_vector_type(4)))  unsigned int u32x4;

#define T_STEPS 256
#define BATCH   64
#define DIN     256
#define DS      512
#define GSZ     (4*DS)            // 2048 gate width
#define NC      10000
#define ROWS    (T_STEPS*BATCH)   // 16384

// ---------------- workspace layout (bytes, all 256-aligned) ----------------
#define OFF_XB    ((size_t)0)            // bf16 [ROWS][DIN]            8,388,608
#define OFF_WXP   ((size_t)8388608)      // bf16 packed B  [DIN x GSZ]  1,048,576
#define OFF_WHP   ((size_t)9437184)      // bf16 packed B  [DS  x GSZ]  2,097,152
#define OFF_WOP   ((size_t)11534336)     // bf16 packed B  [DS  x NC]  10,240,000
#define OFF_HSB   ((size_t)21774336)     // bf16 [ROWS][DS]            16,777,216
#define OFF_HB    ((size_t)38551552)     // bf16 [BATCH][DS]               65,536
#define OFF_CC    ((size_t)38617088)     // f32  [BATCH][DS]              131,072
#define OFF_Z     ((size_t)38748160)     // f32  [BATCH][GSZ]             524,288
#define OFF_RL    ((size_t)39272448)     // f32  [ROWS]                    65,536
#define OFF_SYNC  ((size_t)39337984)     // u32  barrier counters             256

// ---------------- helpers ----------------
__device__ __forceinline__ unsigned short f2bf(float f) {
  unsigned u = __builtin_bit_cast(unsigned, f);
  u += 0x7FFFu + ((u >> 16) & 1u);            // round-to-nearest-even
  return (unsigned short)(u >> 16);
}
__device__ __forceinline__ float sigm(float x) { return 1.f / (1.f + __expf(-x)); }
// branch-free tanh; saturates to +/-1 for large |x|
__device__ __forceinline__ float ftanh(float x) {
  float e = __expf(2.f * x);
  return 1.f - 2.f / (e + 1.f);
}

union Frag { u32x4 u[2]; v16bf v; };

// A fragment (16x32 bf16) from a row-major bf16 matrix (global or LDS pointer)
__device__ __forceinline__ v16bf load_a(const unsigned short* base, int ld,
                                        int row0, int k0, int lane) {
  const int l  = lane & 15;
  const int hi = (lane >> 4) & 1;
  const unsigned short* p = base + (size_t)(row0 + l) * ld + k0 + hi * 8;
  Frag f;
  f.u[0] = *reinterpret_cast<const u32x4*>(p);        // K = k0(+8)   .. +7
  f.u[1] = *reinterpret_cast<const u32x4*>(p + 16);   // K = k0+16(+8).. +7
  return f.v;
}

// B fragment (32x16 bf16) from pre-packed tiles: 32 contiguous bytes per lane
__device__ __forceinline__ v16bf load_b(const unsigned short* packed, int tile, int lane) {
  const unsigned short* p = packed + (size_t)tile * 512 + lane * 16;
  Frag f;
  f.u[0] = reinterpret_cast<const u32x4*>(p)[0];
  f.u[1] = reinterpret_cast<const u32x4*>(p)[1];
  return f.v;
}

__device__ __forceinline__ v8f wmma_bf16(v16bf a, v16bf b, v8f c) {
  return __builtin_amdgcn_wmma_f32_16x16x32_bf16(false, a, false, b, (short)0, c, false, false);
}

// CDNA5 async global->LDS copy of 16 bytes (per-lane addresses, ASYNCcnt)
__device__ __forceinline__ void async_g2l_b128(unsigned lds_addr, const void* gaddr) {
  asm volatile("global_load_async_to_lds_b128 %0, %1, off"
               :: "v"(lds_addr), "v"((unsigned long long)(size_t)gaddr) : "memory");
}
__device__ __forceinline__ void wait_async0() {
  asm volatile("s_wait_asynccnt 0x0" ::: "memory");
}

// device-scope grid barrier (persistent kernel; 64 resident blocks)
__device__ __forceinline__ void grid_barrier(unsigned* cnt, unsigned* gen, unsigned nblk) {
  __syncthreads();
  if (threadIdx.x == 0) {
    unsigned g   = __hip_atomic_load(gen, __ATOMIC_RELAXED, __HIP_MEMORY_SCOPE_AGENT);
    unsigned old = __hip_atomic_fetch_add(cnt, 1u, __ATOMIC_ACQ_REL, __HIP_MEMORY_SCOPE_AGENT);
    if (old == nblk - 1u) {
      __hip_atomic_store(cnt, 0u, __ATOMIC_RELAXED, __HIP_MEMORY_SCOPE_AGENT);
      __hip_atomic_fetch_add(gen, 1u, __ATOMIC_ACQ_REL, __HIP_MEMORY_SCOPE_AGENT);
    } else {
      while (__hip_atomic_load(gen, __ATOMIC_ACQUIRE, __HIP_MEMORY_SCOPE_AGENT) == g)
        __builtin_amdgcn_s_sleep(2);
    }
  }
  __syncthreads();
}

// ---------------- setup kernels ----------------
__global__ void k_init(const float* c0, const float* h0, float* ccur,
                       unsigned short* hb, unsigned* sync) {
  const int n = BATCH * DS;
  for (int i = blockIdx.x * blockDim.x + threadIdx.x; i < n; i += gridDim.x * blockDim.x) {
    ccur[i] = c0[i];
    hb[i]   = f2bf(h0[i]);
  }
  if (blockIdx.x == 0 && threadIdx.x < 8) sync[threadIdx.x] = 0u;
}

__global__ void k_cvt(const float* __restrict__ src, unsigned short* __restrict__ dst, int n) {
  for (int i = blockIdx.x * blockDim.x + threadIdx.x; i < n; i += gridDim.x * blockDim.x)
    dst[i] = f2bf(src[i]);
}

// pack row-major [K][N] fp32 weights into B-fragment tile layout (bf16)
__global__ void k_packB(const float* __restrict__ src, unsigned short* __restrict__ dst,
                        int K, int N) {
  const int ntn = N >> 4;
  const int total = K * N;
  for (int idx = blockIdx.x * blockDim.x + threadIdx.x; idx < total;
       idx += gridDim.x * blockDim.x) {
    int tile = idx >> 9;
    int r    = idx & 511;
    int lane = r >> 4, e = r & 15;
    int kt = tile / ntn, nt = tile - kt * ntn;
    int n  = nt * 16 + (lane & 15);
    int k  = kt * 32 + ((lane & 16) ? 8 : 0) + (e & 7) + ((e & 8) ? 16 : 0);
    dst[idx] = f2bf(src[(size_t)k * N + n]);
  }
}

// ---------------- persistent LSTM recurrence ----------------
// 64 blocks x 256 threads = 512 waves = 4(M) x 128(N) tiles of z per step.
// B fragments register-resident; the block's shared A tile (x_t 16x256 +
// h 16x512, 24KB) is async-staged to LDS each step, WMMAs feed from ds_load.
__global__ void __launch_bounds__(256) k_lstm(
    const unsigned short* __restrict__ xb,  const unsigned short* __restrict__ wxp,
    const unsigned short* __restrict__ whp, const float* __restrict__ bias,
    unsigned short* hb, float* ccur, float* zbuf, unsigned short* hsb, unsigned* sync)
{
  __shared__ __align__(1024) unsigned short smem[12288];  // [0,8KB)=x 16x256, [8KB,24KB)=h 16x512
  const int tid  = threadIdx.x;
  const int lane = tid & 31;
  const int gw   = blockIdx.x * 8 + (tid >> 5);           // 0..511
  const int mt   = gw >> 7;                                // 0..3 (same for all waves in block)
  const int nt   = gw & 127;                               // 0..127
  const int hrow = mt * 16;
  const int gtid = blockIdx.x * blockDim.x + tid;
  const int nthr = gridDim.x * blockDim.x;
  unsigned* cnt = sync;
  unsigned* gen = sync + 1;
  const unsigned lds_base = (unsigned)(size_t)(&smem[0]);
  const unsigned short* xs = &smem[0];
  const unsigned short* hs = &smem[4096];

  // t-invariant B fragments: 8 from Wx, 16 from Wh (192 VGPRs, loaded once)
  v16bf bf[24];
  #pragma unroll
  for (int k = 0; k < 8; ++k)  bf[k]     = load_b(wxp, k * (GSZ / 16) + nt, lane);
  #pragma unroll
  for (int k = 0; k < 16; ++k) bf[8 + k] = load_b(whp, k * (GSZ / 16) + nt, lane);

  const int colZ = nt * 16 + (lane & 15);
  const int r0Z  = mt * 16 + ((lane & 16) ? 8 : 0);
  const float bb = bias[colZ];

  for (int t = 0; t < T_STEPS; ++t) {
    // -------- async stage: x_t tile (512 chunks) + h tile (1024 chunks) --------
    const int xrow0 = t * BATCH + mt * 16;
    for (int c = tid; c < 1536; c += 256) {
      if (c < 512) {
        const int row = c >> 5;                       // 32 x 16B chunks per x row
        async_g2l_b128(lds_base + c * 16,
                       xb + (size_t)(xrow0 + row) * DIN + (c & 31) * 8);
      } else {
        const int cc = c - 512;
        const int row = cc >> 6;                      // 64 x 16B chunks per h row
        async_g2l_b128(lds_base + 8192 + cc * 16,
                       hb + (size_t)(hrow + row) * DS + (cc & 63) * 8);
      }
    }
    wait_async0();
    __syncthreads();

    // -------- stage A compute: z = x_t @ Wx + h @ Wh + b (A from LDS) --------
    v8f acc0 = {}, acc1 = {};
    v16bf fr[24];
    fr[0] = load_a(xs, DIN, 0, 0, lane);
    fr[1] = load_a(xs, DIN, 0, 32, lane);
    #pragma unroll
    for (int k = 0; k < 24; ++k) {
      if (k + 2 < 24) {
        const int kp = k + 2;
        fr[kp] = (kp < 8) ? load_a(xs, DIN, 0, kp * 32, lane)
                          : load_a(hs, DS, 0, (kp - 8) * 32, lane);
      }
      if (k & 1) acc1 = wmma_bf16(fr[k], bf[k], acc1);
      else       acc0 = wmma_bf16(fr[k], bf[k], acc0);
    }
    #pragma unroll
    for (int v = 0; v < 8; ++v)
      zbuf[(r0Z + v) * GSZ + colZ] = acc0[v] + acc1[v] + bb;

    grid_barrier(cnt, gen, gridDim.x);

    // -------- stage B: gates + state update --------
    for (int j = gtid; j < BATCH * DS; j += nthr) {
      const int bi = j >> 9;       // / DS
      const int s  = j & (DS - 1);
      const float* zr = zbuf + bi * GSZ;
      float zi = zr[s], zf = zr[DS + s], zg = zr[2 * DS + s], zo = zr[3 * DS + s];
      float cn = sigm(zf) * ccur[j] + sigm(zi) * ftanh(zg);
      float hn = sigm(zo) * ftanh(cn);
      ccur[j] = cn;
      unsigned short hv = f2bf(hn);
      hb[j] = hv;
      hsb[(size_t)t * BATCH * DS + j] = hv;
    }
    grid_barrier(cnt, gen, gridDim.x);
  }
}

// ---------------- head GEMM: logits = hs @ Wo + bo ----------------
// Block = 64(M) x 256(N): A tile (64x512 bf16, 64KB) async-staged to LDS once;
// each wave computes a 64M x 32N job (8 WMMAs per K-step, B from global/L2).
template <bool FULL>
__device__ __forceinline__ void head_job(const unsigned short* __restrict__ aLds,
                                         const unsigned short* __restrict__ wop,
                                         const float* __restrict__ bo,
                                         float* __restrict__ out,
                                         int row0, int nt0, int lane) {
  const int NT = NC / 16;  // 625
  v8f c00 = {}, c10 = {}, c20 = {}, c30 = {};
  v8f c01 = {}, c11 = {}, c21 = {}, c31 = {};
  #pragma unroll
  for (int kt = 0; kt < DS / 32; ++kt) {
    v16bf b0 = load_b(wop, kt * NT + nt0, lane);
    v16bf a0 = load_a(aLds, DS, 0,  kt * 32, lane);
    v16bf a1 = load_a(aLds, DS, 16, kt * 32, lane);
    v16bf a2 = load_a(aLds, DS, 32, kt * 32, lane);
    v16bf a3 = load_a(aLds, DS, 48, kt * 32, lane);
    c00 = wmma_bf16(a0, b0, c00);
    c10 = wmma_bf16(a1, b0, c10);
    c20 = wmma_bf16(a2, b0, c20);
    c30 = wmma_bf16(a3, b0, c30);
    if (FULL) {
      v16bf b1 = load_b(wop, kt * NT + nt0 + 1, lane);
      c01 = wmma_bf16(a0, b1, c01);
      c11 = wmma_bf16(a1, b1, c11);
      c21 = wmma_bf16(a2, b1, c21);
      c31 = wmma_bf16(a3, b1, c31);
    }
  }
  const int col0 = nt0 * 16 + (lane & 15);
  const int rb   = row0 + ((lane & 16) ? 8 : 0);
  const float b0v = bo[col0];
  const float b1v = FULL ? bo[col0 + 16] : 0.f;
  v8f cs0[4] = {c00, c10, c20, c30};
  v8f cs1[4] = {c01, c11, c21, c31};
  #pragma unroll
  for (int m = 0; m < 4; ++m) {
    #pragma unroll
    for (int v = 0; v < 8; ++v) {
      size_t r = (size_t)(rb + m * 16 + v) * NC;
      out[r + col0] = cs0[m][v] + b0v;
      if (FULL) out[r + col0 + 16] = cs1[m][v] + b1v;
    }
  }
}

#define HEAD_NGG 40   // ceil(313 N-groups / 8 waves)

__global__ void __launch_bounds__(256) k_head(
    const unsigned short* __restrict__ hsb, const unsigned short* __restrict__ wop,
    const float* __restrict__ bo, float* __restrict__ out)
{
  __shared__ __align__(1024) unsigned short smh[32768];  // 64 rows x 512 K bf16 = 64KB
  const int tid  = threadIdx.x;
  const int lane = tid & 31;
  const int mg   = blockIdx.x & 255;        // 256 M-groups of 64 rows
  const int ngg  = blockIdx.x >> 8;         // 40 N-octets
  const int row0 = mg * 64;
  const int NT = NC / 16;                   // 625
  const int NG = (NT + 1) / 2;              // 313

  // async stage the 64x512 A tile (4096 x 16B chunks, 16 per thread)
  const unsigned lds_base = (unsigned)(size_t)(&smh[0]);
  for (int c = tid; c < 4096; c += 256) {
    const int row = c >> 6;                 // 64 x 16B chunks per row
    async_g2l_b128(lds_base + c * 16,
                   hsb + (size_t)(row0 + row) * DS + (c & 63) * 8);
  }
  wait_async0();
  __syncthreads();

  const int ng = ngg * 8 + (tid >> 5);
  if (ng >= NG) return;
  const int nt0 = ng * 2;
  if (nt0 + 1 < NT) head_job<true >(smh, wop, bo, out, row0, nt0, lane);
  else              head_job<false>(smh, wop, bo, out, row0, nt0, lane);
}

// ---------------- in-place softmax + per-row loss ----------------
__global__ void __launch_bounds__(256) k_softmax(const float* __restrict__ labels,
                                                 float* __restrict__ out,
                                                 float* __restrict__ rowloss)
{
  __shared__ float red[256];
  const int row = blockIdx.x;
  float* x = out + (size_t)row * NC;
  const float* lb = labels + (size_t)row * NC;
  const int tid = threadIdx.x;

  float m = -3.4e38f;
  for (int i = tid; i < NC; i += 256) m = fmaxf(m, x[i]);
  red[tid] = m; __syncthreads();
  for (int s = 128; s > 0; s >>= 1) { if (tid < s) red[tid] = fmaxf(red[tid], red[tid + s]); __syncthreads(); }
  m = red[0]; __syncthreads();

  float se = 0.f, dot = 0.f;
  for (int i = tid; i < NC; i += 256) { float v = x[i]; se += __expf(v - m); dot += lb[i] * v; }
  red[tid] = se; __syncthreads();
  for (int s = 128; s > 0; s >>= 1) { if (tid < s) red[tid] += red[tid + s]; __syncthreads(); }
  se = red[0]; __syncthreads();
  red[tid] = dot; __syncthreads();
  for (int s = 128; s > 0; s >>= 1) { if (tid < s) red[tid] += red[tid + s]; __syncthreads(); }
  dot = red[0];

  const float lse = m + __logf(se);
  for (int i = tid; i < NC; i += 256) x[i] = __expf(x[i] - lse);
  if (tid == 0) rowloss[row] = lse - dot;   // -sum(labels*log_p), labels sum to 1
}

__global__ void k_loss(const float* __restrict__ rowloss, float* __restrict__ out_loss) {
  __shared__ float red[256];
  const int tid = threadIdx.x;
  float s = 0.f;
  for (int i = tid; i < ROWS; i += 256) s += rowloss[i];
  red[tid] = s; __syncthreads();
  for (int k = 128; k > 0; k >>= 1) { if (tid < k) red[tid] += red[tid + k]; __syncthreads(); }
  if (tid == 0) *out_loss = red[0] / (float)ROWS;
}

// ---------------- launch ----------------
extern "C" void kernel_launch(void* const* d_in, const int* in_sizes, int n_in,
                              void* d_out, int out_size, void* d_ws, size_t ws_size,
                              hipStream_t stream) {
  const float* inputs = (const float*)d_in[0];
  const float* labels = (const float*)d_in[1];
  const float* c0     = (const float*)d_in[2];
  const float* h0     = (const float*)d_in[3];
  const float* Wx     = (const float*)d_in[4];
  const float* Wh     = (const float*)d_in[5];
  const float* bias   = (const float*)d_in[6];
  const float* Wo     = (const float*)d_in[7];
  const float* bo     = (const float*)d_in[8];

  char* w = (char*)d_ws;
  unsigned short* xb   = (unsigned short*)(w + OFF_XB);
  unsigned short* wxp  = (unsigned short*)(w + OFF_WXP);
  unsigned short* whp  = (unsigned short*)(w + OFF_WHP);
  unsigned short* wop  = (unsigned short*)(w + OFF_WOP);
  unsigned short* hsb  = (unsigned short*)(w + OFF_HSB);
  unsigned short* hb   = (unsigned short*)(w + OFF_HB);
  float*          ccur = (float*)(w + OFF_CC);
  float*          zbuf = (float*)(w + OFF_Z);
  float*          rl   = (float*)(w + OFF_RL);
  unsigned*       sync = (unsigned*)(w + OFF_SYNC);
  float*          out  = (float*)d_out;

  k_init   <<<128, 256, 0, stream>>>(c0, h0, ccur, hb, sync);
  k_cvt    <<<4096, 256, 0, stream>>>(inputs, xb, ROWS * DIN);
  k_packB  <<<2048, 256, 0, stream>>>(Wx, wxp, DIN, GSZ);
  k_packB  <<<4096, 256, 0, stream>>>(Wh, whp, DS, GSZ);
  k_packB  <<<8192, 256, 0, stream>>>(Wo, wop, DS, NC);
  k_lstm   <<<64, 256, 0, stream>>>(xb, wxp, whp, bias, hb, ccur, zbuf, hsb, sync);
  k_head   <<<256 * HEAD_NGG, 256, 0, stream>>>(hsb, wop, bo, out);
  k_softmax<<<ROWS, 256, 0, stream>>>(labels, out, rl);
  k_loss   <<<1, 256, 0, stream>>>(rl, out + (size_t)ROWS * NC);
}
